// GeoFeatGenerator_63213328663283
// MI455X (gfx1250) — compile-verified, matrix operand order checked by reading, not run on hardware
//
#include <hip/hip_runtime.h>
#include <hip/hip_bf16.h>

typedef __attribute__((ext_vector_type(16))) _Float16 v16h;
typedef __attribute__((ext_vector_type(8)))  float    v8f;

#define NEG_INF (-3.402823e38f)

// ---- optional CDNA5 async global->LDS copy path (guarded; falls back to plain copy)
#ifdef __has_builtin
#if __has_builtin(__builtin_amdgcn_global_load_async_to_lds_b32)
#define ATHENA_ASYNC_LDS 1
#endif
#endif

typedef __attribute__((address_space(1))) int as1_int;
typedef __attribute__((address_space(3))) int as3_int;

__device__ __forceinline__ void g2l_b32(void* ldsDst, const void* gSrc) {
#ifdef ATHENA_ASYNC_LDS
  // flat shared address low 32 bits == LDS byte offset (ISA 10.2 aperture rules)
  __builtin_amdgcn_global_load_async_to_lds_b32(
      (as1_int*)(uintptr_t)gSrc,
      (as3_int*)(uint32_t)(uintptr_t)ldsDst, 0, 0);
#else
  *(uint32_t*)ldsDst = *(const uint32_t*)gSrc;
#endif
}

__device__ __forceinline__ void g2l_fence() {
#ifdef ATHENA_ASYNC_LDS
#if __has_builtin(__builtin_amdgcn_s_wait_asynccnt)
  __builtin_amdgcn_s_wait_asynccnt(0);
#else
  asm volatile("s_wait_asynccnt 0" ::: "memory");
#endif
#endif
}

// K index inside a 16x32 f16 A/B fragment for half-slot h (0..15), hi = lane>>4
__device__ __forceinline__ int kmap(int h, int hi) {
  return (h < 8 ? h : h + 8) + (hi << 3);
}

__device__ __forceinline__ v8f wmma32(v16h a, v16h b, v8f c) {
  return __builtin_amdgcn_wmma_f32_16x16x32_f16(false, a, false, b, (short)0, c, false, false);
}

// ---------------------------------------------------------------- embeddings
__global__ __launch_bounds__(256) void embed_kernel(
    const int* __restrict__ x, const float* __restrict__ occ_emb,
    const float* __restrict__ lvl_emb, const float* __restrict__ oct_emb,
    float* __restrict__ xf, int BN) {
  int p = blockIdx.x * 256 + threadIdx.x;
  if (p >= BN) return;
  const int* xr = x + (size_t)p * 11;
  float* o = xf + (size_t)p * 80;
  #pragma unroll
  for (int g = 0; g < 3; ++g) {
    int id = xr[2 + 3 * g];
    #pragma unroll
    for (int j = 0; j < 16; ++j) o[g * 16 + j] = occ_emb[id * 16 + j];
  }
  #pragma unroll
  for (int g = 0; g < 4; ++g) {
    int id = xr[3 * g];
    #pragma unroll
    for (int j = 0; j < 4; ++j) o[48 + g * 4 + j] = lvl_emb[id * 4 + j];
  }
  #pragma unroll
  for (int g = 0; g < 4; ++g) {
    int id = xr[1 + 3 * g];
    #pragma unroll
    for (int j = 0; j < 4; ++j) o[64 + g * 4 + j] = oct_emb[id * 4 + j];
  }
}

// ------------------------------------------- pack pos (B,3,N) -> f16 rows + |x|^2
__global__ __launch_bounds__(256) void pack_pos_kernel(
    const float* __restrict__ pos, _Float16* __restrict__ fh,
    float* __restrict__ sq, int N, int BN) {
  int p = blockIdx.x * 256 + threadIdx.x;
  if (p >= BN) return;
  int b = p / N, n = p - b * N;
  float s = 0.f;
  _Float16* row = fh + (size_t)p * 32;
  #pragma unroll
  for (int c = 0; c < 3; ++c) {
    float v = pos[((size_t)b * 3 + c) * N + n];
    s += v * v;
    row[c] = (_Float16)v;
  }
  #pragma unroll
  for (int c = 3; c < 32; ++c) row[c] = (_Float16)0.f;
  sq[p] = s;
}

// -------------------------- pack concat of two point-major f32 srcs -> f16 + |x|^2
__global__ __launch_bounds__(256) void pack2_kernel(
    const float* __restrict__ a, int aS, int aO, int Ca,
    const float* __restrict__ bsrc, int bS, int bO, int Cb,
    _Float16* __restrict__ fh, int Kpad, float* __restrict__ sq, int BN) {
  int p = blockIdx.x * 256 + threadIdx.x;
  if (p >= BN) return;
  float s = 0.f;
  _Float16* row = fh + (size_t)p * Kpad;
  int c = 0;
  for (int i = 0; i < Ca; ++i) { float v = a[(size_t)p * aS + aO + i]; s += v * v; row[c++] = (_Float16)v; }
  for (int i = 0; i < Cb; ++i) { float v = bsrc[(size_t)p * bS + bO + i]; s += v * v; row[c++] = (_Float16)v; }
  for (; c < Kpad; ++c) row[c] = (_Float16)0.f;
  sq[p] = s;
}

// -------------------- convert + zero-pad conv weights (O, 2C) f32 -> (O, K2pad) f16
__global__ __launch_bounds__(256) void convw_kernel(
    const float* __restrict__ w, _Float16* __restrict__ wh,
    int C2, int K2pad, int Cout) {
  int e = blockIdx.x * 256 + threadIdx.x;
  if (e >= Cout * K2pad) return;
  int o = e / K2pad, c = e - o * K2pad;
  wh[e] = (c < C2) ? (_Float16)w[(size_t)o * C2 + c] : (_Float16)0.f;
}

// ------------------------------------------------ WMMA gram + top-20 neighbors
template <int NCHUNK>
__global__ __launch_bounds__(256) void knn_kernel(
    const _Float16* __restrict__ feat, const float* __restrict__ sq,
    int* __restrict__ knn_idx, int N) {
  const int KPAD = NCHUNK * 32;
  const int tid = threadIdx.x;
  const int wave = tid >> 5, lane = tid & 31, lo16 = lane & 15, hi = lane >> 4;
  const int b = blockIdx.y;
  const int rblk = blockIdx.x * 128;

  __shared__ float tile_s[8][16][16];
  __shared__ float sqrow_s[128];
  __shared__ float md_s[8][32][20];   // merge dists; aliased as Bt during column loop
  __shared__ int   mi_s[8][32][20];   // merge idx
  _Float16* Bt = (_Float16*)&md_s[0][0][0];  // 16*KPAD f16 <= 6KB, fits in 20KB

  if (tid < 128) sqrow_s[tid] = sq[(size_t)b * N + rblk + tid];

  const int r0 = rblk + wave * 16;
  const _Float16* rowp = feat + ((size_t)b * N + r0 + lo16) * KPAD;

  v16h afr[NCHUNK];
  #pragma unroll
  for (int kc = 0; kc < NCHUNK; ++kc) {
    #pragma unroll
    for (int h = 0; h < 16; ++h) afr[kc][h] = rowp[kc * 32 + kmap(h, hi)];
  }

  float bd[20]; int bi[20];
  #pragma unroll
  for (int j = 0; j < 20; ++j) { bd[j] = NEG_INF; bi[j] = 0; }

  const int ntiles = N >> 4;
  const int dwords = (16 * KPAD) >> 1;  // f16 pairs in a column tile
  for (int ct = 0; ct < ntiles; ++ct) {
    const int c0 = ct << 4;
    __syncthreads();  // previous iteration fully consumed Bt/tile_s
    // stage the shared 16 x KPAD column tile once per block (async -> LDS)
    for (int e = tid; e < dwords; e += 256) {
      const int f = e << 1;
      const int row = f / KPAD, col = f - row * KPAD;
      g2l_b32((uint32_t*)Bt + e, feat + ((size_t)b * N + c0 + row) * KPAD + col);
    }
    if (ct + 1 < ntiles)
      __builtin_prefetch(feat + ((size_t)b * N + c0 + 16 + lo16) * KPAD, 0, 0);
    g2l_fence();
    __syncthreads();  // Bt visible to all waves

    const _Float16* btp = Bt + (size_t)lo16 * KPAD;
    v8f acc = {};
    #pragma unroll
    for (int kc = 0; kc < NCHUNK; ++kc) {
      v16h bf;
      #pragma unroll
      for (int h = 0; h < 16; ++h) bf[h] = btp[kc * 32 + kmap(h, hi)];
      acc = wmma32(afr[kc], bf, acc);
    }
    const float sqc = sq[(size_t)b * N + c0 + lo16];
    #pragma unroll
    for (int r = 0; r < 8; ++r) {
      const int m = r + (hi << 3);
      tile_s[wave][m][lo16] = 2.0f * acc[r] - sqrow_s[wave * 16 + m] - sqc;
    }
    __syncthreads();  // tile_s transposed view ready
    // lane owns row lo16, column-half hi: ripple-insert 8 candidates
    #pragma unroll
    for (int j = 0; j < 8; ++j) {
      const int nn = (hi << 3) + j;
      float v = tile_s[wave][lo16][nn];
      int vi = c0 + nn;
      #pragma unroll
      for (int t = 0; t < 20; ++t) {
        if (v > bd[t]) { float tv = bd[t]; int ti = bi[t]; bd[t] = v; bi[t] = vi; v = tv; vi = ti; }
      }
    }
  }

  __syncthreads();  // done with Bt alias; switch region to merge arrays
  #pragma unroll
  for (int j = 0; j < 20; ++j) { md_s[wave][lane][j] = bd[j]; mi_s[wave][lane][j] = bi[j]; }
  __syncthreads();
  if (lane < 16) {
    float fd[20]; int fi[20];
    #pragma unroll
    for (int j = 0; j < 20; ++j) { fd[j] = NEG_INF; fi[j] = 0; }
    #pragma unroll
    for (int half = 0; half < 2; ++half) {
      #pragma unroll
      for (int j = 0; j < 20; ++j) {
        float v = md_s[wave][lane + half * 16][j];
        int vi = mi_s[wave][lane + half * 16][j];
        #pragma unroll
        for (int t = 0; t < 20; ++t) {
          if (v > fd[t]) { float tv = fd[t]; int ti = fi[t]; fd[t] = v; fi[t] = vi; v = tv; vi = ti; }
        }
      }
    }
    int* op = knn_idx + ((size_t)b * N + r0 + lane) * 20;
    #pragma unroll
    for (int j = 0; j < 20; ++j) op[j] = fi[j];
  }
}

// ------------------ edge conv: gather, WMMA GEMM, BN + LeakyReLU(0.2), max over k
__global__ __launch_bounds__(256) void edgeconv_kernel(
    const _Float16* __restrict__ feat, int featStride,
    const int* __restrict__ knn_idx, const _Float16* __restrict__ whG,
    const float* __restrict__ gamma, const float* __restrict__ beta,
    const float* __restrict__ mean, const float* __restrict__ var,
    float* __restrict__ out1, int s1, int o1,
    float* __restrict__ out2, int s2, int o2,
    int N, int Cin, int Cout, int K2pad) {
  extern __shared__ char smem[];
  _Float16* Wh = (_Float16*)smem;
  _Float16* Af = Wh + (size_t)Cout * K2pad;
  int* nbr_s = (int*)(Af + (size_t)16 * K2pad);

  const int tid = threadIdx.x;
  const int wave = tid >> 5, lane = tid & 31, lo16 = lane & 15, hi = lane >> 4;
  const int p0 = blockIdx.x * 16;
  const int b = p0 / N;
  const int C2 = 2 * Cin;
  const int nchunk = K2pad >> 5;
  const int ctiles = Cout >> 4;

  // stage pre-converted padded f16 weights via async copy
  const int wdw = (Cout * K2pad) >> 1;
  for (int e = tid; e < wdw; e += 256)
    g2l_b32((uint32_t*)Wh + e, (const uint32_t*)whG + e);
  g2l_fence();

  float mx[2][8];
  #pragma unroll
  for (int s = 0; s < 2; ++s)
    #pragma unroll
    for (int r = 0; r < 8; ++r) mx[s][r] = NEG_INF;

  for (int kk = 0; kk < 20; ++kk) {
    __syncthreads();
    if (tid < 16) nbr_s[tid] = knn_idx[((size_t)p0 + tid) * 20 + kk];
    __syncthreads();
    for (int e = tid; e < 16 * K2pad; e += 256) {
      int p = e / K2pad, c = e - p * K2pad;
      _Float16 v = (_Float16)0.f;
      if (c < Cin) {
        float ctr = (float)feat[((size_t)p0 + p) * featStride + c];
        float nb  = (float)feat[((size_t)b * N + nbr_s[p]) * featStride + c];
        v = (_Float16)(nb - ctr);
      } else if (c < C2) {
        v = feat[((size_t)p0 + p) * featStride + (c - Cin)];
      }
      Af[e] = v;
    }
    __syncthreads();
    for (int t = wave; t < ctiles; t += 8) {
      const int o = t * 16 + lo16;
      v8f acc = {};
      for (int kc = 0; kc < nchunk; ++kc) {
        v16h a, bf;
        #pragma unroll
        for (int h = 0; h < 16; ++h) {
          const int k = kc * 32 + kmap(h, hi);
          a[h]  = Af[(size_t)lo16 * K2pad + k];
          bf[h] = Wh[(size_t)o * K2pad + k];
        }
        acc = wmma32(a, bf, acc);
      }
      const float sc = gamma[o] * rsqrtf(var[o] + 1e-5f);
      const float sh = beta[o] - mean[o] * sc;
      const int slot = (t - wave) >> 3;
      #pragma unroll
      for (int r = 0; r < 8; ++r) {
        float y = acc[r] * sc + sh;
        y = y > 0.f ? y : 0.2f * y;
        mx[slot][r] = fmaxf(mx[slot][r], y);
      }
    }
  }
  __syncthreads();
  for (int t = wave; t < ctiles; t += 8) {
    const int o = t * 16 + lo16;
    const int slot = (t - wave) >> 3;
    #pragma unroll
    for (int r = 0; r < 8; ++r) {
      const int p = p0 + r + (hi << 3);
      out1[(size_t)p * s1 + o1 + o] = mx[slot][r];
      if (out2) out2[(size_t)p * s2 + o2 + o] = mx[slot][r];
    }
  }
}

// ---------------------------- generic point-wise GEMM + bias + optional LeakyReLU
__global__ __launch_bounds__(256) void gemm_bias_act_kernel(
    const float* __restrict__ in, int inS, int inO, int Cin,
    const float* __restrict__ w, const float* __restrict__ bias, int Cout,
    float* __restrict__ out, int outS, int outO, int act, float slope) {
  __shared__ float Ahf[16 * 512];  // f32 A tile, async-staged raw, converted on use
  const int tid = threadIdx.x;
  const int wave = tid >> 5, lane = tid & 31, lo16 = lane & 15, hi = lane >> 4;
  const int p0 = blockIdx.x * 16;
  const int Kpad = (Cin + 31) & ~31;
  const int nchunk = Kpad >> 5;
  const int ctiles = Cout >> 4;

  for (int e = tid; e < 16 * Kpad; e += 256) {
    int p = e / Kpad, c = e - p * Kpad;
    if (c < Cin) g2l_b32(Ahf + e, in + (size_t)(p0 + p) * inS + inO + c);
    else Ahf[e] = 0.f;
  }
  g2l_fence();
  __syncthreads();

  for (int t = wave; t < ctiles; t += 8) {
    const int o = t * 16 + lo16;
    v8f acc = {};
    for (int kc = 0; kc < nchunk; ++kc) {
      v16h a, bf;
      #pragma unroll
      for (int h = 0; h < 16; ++h) {
        const int k = kc * 32 + kmap(h, hi);
        a[h]  = (_Float16)Ahf[(size_t)lo16 * Kpad + k];
        bf[h] = (k < Cin) ? (_Float16)w[(size_t)k * Cout + o] : (_Float16)0.f;
      }
      acc = wmma32(a, bf, acc);
    }
    const float bo = bias[o];
    #pragma unroll
    for (int r = 0; r < 8; ++r) {
      float y = acc[r] + bo;
      if (act) y = y > 0.f ? y : slope * y;
      out[(size_t)(p0 + r + (hi << 3)) * outS + outO + o] = y;
    }
  }
}

// --------------------------------------------------------------------- launcher
extern "C" void kernel_launch(void* const* d_in, const int* in_sizes, int n_in,
                              void* d_out, int out_size, void* d_ws, size_t ws_size,
                              hipStream_t stream) {
  const int B = 4, N = 4096, BN = B * N;
  (void)in_sizes; (void)n_in; (void)out_size; (void)ws_size;

  const int*   x       = (const int*)d_in[0];
  const float* pos     = (const float*)d_in[1];
  const float* occ_emb = (const float*)d_in[2];
  const float* lvl_emb = (const float*)d_in[3];
  const float* oct_emb = (const float*)d_in[4];
  const float* c1w = (const float*)d_in[5];
  const float *c1g = (const float*)d_in[6], *c1b = (const float*)d_in[7],
              *c1m = (const float*)d_in[8], *c1v = (const float*)d_in[9];
  const float* c2w = (const float*)d_in[10];
  const float *c2g = (const float*)d_in[11], *c2b = (const float*)d_in[12],
              *c2m = (const float*)d_in[13], *c2v = (const float*)d_in[14];
  const float* c3w = (const float*)d_in[15];
  const float *c3g = (const float*)d_in[16], *c3b = (const float*)d_in[17],
              *c3m = (const float*)d_in[18], *c3v = (const float*)d_in[19];
  const float *m2w0 = (const float*)d_in[20], *m2b0 = (const float*)d_in[21],
              *m2w1 = (const float*)d_in[22], *m2b1 = (const float*)d_in[23],
              *m2w2 = (const float*)d_in[24], *m2b2 = (const float*)d_in[25];
  const float *m3w0 = (const float*)d_in[26], *m3b0 = (const float*)d_in[27],
              *m3w1 = (const float*)d_in[28], *m3b1 = (const float*)d_in[29],
              *m3w2 = (const float*)d_in[30], *m3b2 = (const float*)d_in[31];
  const float *e1w0 = (const float*)d_in[32], *e1b0 = (const float*)d_in[33],
              *e1w1 = (const float*)d_in[34], *e1b1 = (const float*)d_in[35],
              *e1w2 = (const float*)d_in[36], *e1b2 = (const float*)d_in[37];
  const float *e2w0 = (const float*)d_in[38], *e2b0 = (const float*)d_in[39],
              *e2w1 = (const float*)d_in[40], *e2b1 = (const float*)d_in[41],
              *e2w2 = (const float*)d_in[42], *e2b2 = (const float*)d_in[43];

  char* wsp = (char*)d_ws;
  auto alloc = [&](size_t bytes) {
    char* p = wsp;
    wsp += (bytes + 255) & ~(size_t)255;
    return p;
  };
  float*    xf     = (float*)alloc((size_t)BN * 80 * 4);
  _Float16* featH  = (_Float16*)alloc((size_t)BN * 192 * 2);
  float*    sq     = (float*)alloc((size_t)BN * 4);
  int*      idx    = (int*)alloc((size_t)BN * 20 * 4);
  float*    posCat = (float*)alloc((size_t)BN * 448 * 4);  // pos1@0, pos2@64, pos3@192
  float*    tA     = (float*)alloc((size_t)BN * 256 * 4);
  float*    tB     = (float*)alloc((size_t)BN * 256 * 4);
  float*    x2     = (float*)alloc((size_t)BN * 64 * 4);
  float*    ecat   = (float*)alloc((size_t)BN * 512 * 4);  // pos3@0, ec@256
  _Float16* wh1    = (_Float16*)alloc((size_t)64 * 32 * 2);
  _Float16* wh2    = (_Float16*)alloc((size_t)128 * 288 * 2);
  _Float16* wh3    = (_Float16*)alloc((size_t)256 * 384 * 2);

  float* outF = (float*)d_out;  // (B, N, 256): x3 @ 0..127, ec @ 128..255

  auto gemm = [&](const float* in, int inS, int inO, int Cin, const float* w,
                  const float* bias, int Cout, float* out, int outS, int outO,
                  int act, float slope) {
    gemm_bias_act_kernel<<<BN / 16, 256, 0, stream>>>(in, inS, inO, Cin, w, bias,
                                                      Cout, out, outS, outO, act, slope);
  };

  // ---- weight pre-conversion (padded f16, consumed by async LDS staging)
  convw_kernel<<<(64 * 32 + 255) / 256, 256, 0, stream>>>(c1w, wh1, 6, 32, 64);
  convw_kernel<<<(128 * 288 + 255) / 256, 256, 0, stream>>>(c2w, wh2, 288, 288, 128);
  convw_kernel<<<(256 * 384 + 255) / 256, 256, 0, stream>>>(c3w, wh3, 384, 384, 256);

  // ---- embeddings + stage 1 features
  embed_kernel<<<(BN + 255) / 256, 256, 0, stream>>>(x, occ_emb, lvl_emb, oct_emb, xf, BN);
  pack_pos_kernel<<<(BN + 255) / 256, 256, 0, stream>>>(pos, featH, sq, N, BN);

  dim3 kgrid(N / 128, B);
  knn_kernel<1><<<kgrid, 256, 0, stream>>>(featH, sq, idx, N);
  size_t sm1 = ((size_t)(64 + 16) * 32) * 2 + 64;
  edgeconv_kernel<<<BN / 16, 256, sm1, stream>>>(featH, 32, idx, wh1, c1g, c1b, c1m, c1v,
                                                 posCat, 448, 0, nullptr, 0, 0, N, 3, 64, 32);

  // ---- mlp2: 80 -> 80 -> 64 -> 64 (slope 0.01, last linear)
  gemm(xf, 80, 0, 80, m2w0, m2b0, 80, tA, 80, 0, 1, 0.01f);
  gemm(tA, 80, 0, 80, m2w1, m2b1, 64, tB, 64, 0, 1, 0.01f);
  gemm(tB, 64, 0, 64, m2w2, m2b2, 64, x2, 64, 0, 0, 0.f);

  // ---- stage 2: feature = [pos1(64), xf(80)] -> 144 (pad 160)
  pack2_kernel<<<(BN + 255) / 256, 256, 0, stream>>>(posCat, 448, 0, 64, xf, 80, 0, 80,
                                                     featH, 160, sq, BN);
  knn_kernel<5><<<kgrid, 256, 0, stream>>>(featH, sq, idx, N);
  size_t sm2 = ((size_t)(128 + 16) * 288) * 2 + 64;
  edgeconv_kernel<<<BN / 16, 256, sm2, stream>>>(featH, 160, idx, wh2, c2g, c2b, c2m, c2v,
                                                 posCat, 448, 64, nullptr, 0, 0, N, 144, 128, 288);

  // ---- mlp3: 64 -> 128 -> 128 -> 128, last layer straight into output cols 0..127
  gemm(x2, 64, 0, 64, m3w0, m3b0, 128, tA, 128, 0, 1, 0.01f);
  gemm(tA, 128, 0, 128, m3w1, m3b1, 128, tB, 128, 0, 1, 0.01f);
  gemm(tB, 128, 0, 128, m3w2, m3b2, 128, outF, 256, 0, 0, 0.f);

  // ---- stage 3: feature = [pos2(128), x2(64)] -> 192
  pack2_kernel<<<(BN + 255) / 256, 256, 0, stream>>>(posCat, 448, 64, 128, x2, 64, 0, 64,
                                                     featH, 192, sq, BN);
  knn_kernel<6><<<kgrid, 256, 0, stream>>>(featH, sq, idx, N);
  size_t sm3 = ((size_t)(256 + 16) * 384) * 2 + 64;
  edgeconv_kernel<<<BN / 16, 256, sm3, stream>>>(featH, 192, idx, wh3, c3g, c3b, c3m, c3v,
                                                 posCat, 448, 192, ecat, 512, 0, N, 192, 256, 384);

  // ---- edge_mlp1: concat(pos1,pos2,pos3)=448 -> 256 -> 256 -> 256 (into ecat @256)
  gemm(posCat, 448, 0, 448, e1w0, e1b0, 256, tA, 256, 0, 1, 0.01f);
  gemm(tA, 256, 0, 256, e1w1, e1b1, 256, tB, 256, 0, 1, 0.01f);
  gemm(tB, 256, 0, 256, e1w2, e1b2, 256, ecat, 512, 256, 0, 0.f);

  // ---- edge_mlp2: concat(pos3, ec)=512 -> 256 -> 256 -> 128, into output cols 128..255
  gemm(ecat, 512, 0, 512, e2w0, e2b0, 256, tA, 256, 0, 1, 0.01f);
  gemm(tA, 256, 0, 256, e2w1, e2b1, 256, tB, 256, 0, 1, 0.01f);
  gemm(tB, 256, 0, 256, e2w2, e2b2, 128, outF, 256, 128, 0, 0.f);
}
